// GAT_3324304687178
// MI455X (gfx1250) — matrix operand order repaired and checked by reading, not compiled, not run on hardware
//
#include <hip/hip_runtime.h>
#include <hip/hip_bf16.h>
#include <stdint.h>

#define N_NODES 50000
#define N_EDGES 800000
#define ET      (N_EDGES + N_NODES)   // edges + self loops
#define F_INPUT 256
#define H_DIM   128
#define C_DIM   40
#define NEG_SLOPE 0.2f
#define BN_EPS 1e-5f
#define ROW_TILES (N_NODES / 16)      // 3125

typedef __attribute__((ext_vector_type(16))) __bf16 v16bf;
typedef __attribute__((ext_vector_type(8)))  float  v8f;

// order-preserving float <-> uint encoding for atomic max
__device__ __forceinline__ unsigned fenc(float f) {
  unsigned u = __float_as_uint(f);
  return (u & 0x80000000u) ? ~u : (u | 0x80000000u);
}
__device__ __forceinline__ float fdec(unsigned u) {
  unsigned b = (u & 0x80000000u) ? (u ^ 0x80000000u) : ~u;
  return __uint_as_float(b);
}

// ---------------------------------------------------------------------------
// Pack W[K x M] (fp32, row-major) into bf16 WMMA B-fragments:
//   Wp[((kc*colTiles + ct)*32 + lane)*16 + e] = W[kc*32 + (lane>>4)*16 + e][ct*16 + (lane&15)]
// M=40 tail zero-filled at pack time -> branch-free GEMM inner loop.
// ---------------------------------------------------------------------------
__global__ void pack_w_bf16(const float* __restrict__ W, __bf16* __restrict__ Wp,
                            int K, int M, int colTiles) {
  int i = blockIdx.x * blockDim.x + threadIdx.x;
  int total = (K >> 5) * colTiles * 512;
  if (i >= total) return;
  int e    = i & 15;
  int lane = (i >> 4) & 31;
  int ct   = (i >> 9) % colTiles;
  int kc   = (i >> 9) / colTiles;
  int k    = kc * 32 + (lane >> 4) * 16 + e;
  int col  = ct * 16 + (lane & 15);
  Wp[i] = (col < M) ? (__bf16)W[(size_t)k * M + col] : (__bf16)0.0f;
}

// ---------------------------------------------------------------------------
// C[N x M] = A[N x K] @ W[K x M]; one wave computes a full 16-row strip across
// all CT column tiles: the A fragment is loaded ONCE per K-chunk and reused by
// CT WMMAs (8x fewer A loads, 8x WMMA density vs tile-per-wave).
// CT is compile-time so acc[CT] stays fully in VGPRs.
// ---------------------------------------------------------------------------
template <int CT>
__global__ void gemm_wmma_strip(const float* __restrict__ A,
                                const __bf16* __restrict__ Wp,
                                float* __restrict__ Cm,
                                int K, int M) {
  const int wave = threadIdx.x >> 5;
  const int lane = threadIdx.x & 31;
  const int half = lane >> 4;
  const int l    = lane & 15;
  const int rowTile = blockIdx.x * (blockDim.x >> 5) + wave;
  if (rowTile >= ROW_TILES) return;
  const int rowBase = rowTile << 4;

  v8f acc[CT];
#pragma unroll
  for (int c = 0; c < CT; ++c) acc[c] = (v8f){};

  const float*  arow  = A  + (size_t)(rowBase + l) * K;
  const __bf16* wbase = Wp + (size_t)lane * 16;

  for (int kb = 0, kc = 0; kb < K; kb += 32, ++kc) {
    v16bf a;
    // A fragment in contiguous k order (matches packed B, not AMD's table order): lane<16 -> K {0..15}, lane>=16 -> K {16..31}
#pragma unroll
    for (int i = 0; i < 8; ++i) {
      a[i]     = (__bf16)arow[kb + half * 16 + i];
      a[8 + i] = (__bf16)arow[kb + half * 16 + 8 + i];
    }
    const __bf16* wk = wbase + (size_t)kc * CT * 512;
#pragma unroll
    for (int c = 0; c < CT; ++c) {
      v16bf b = *(const v16bf*)(wk + (size_t)c * 512);  // contiguous 32B fragment
      acc[c] = __builtin_amdgcn_wmma_f32_16x16x32_bf16(
          /*neg_a=*/false, a, /*neg_b=*/false, b,
          /*c_mod=*/(short)0, acc[c], /*reuse_a=*/false, /*reuse_b=*/false);
    }
  }

#pragma unroll
  for (int c = 0; c < CT; ++c) {
    int col = c * 16 + l;
    if (col < M) {
#pragma unroll
      for (int v = 0; v < 8; ++v)   // C/D layout: VGPR v -> row v (+8 for hi lanes)
        Cm[(size_t)(rowBase + half * 8 + v) * M + col] = acc[c][v];
    }
  }
}

// ---------------------------------------------------------------------------
// per-node attention projections: vs[n] = h[n].a_src, vd[n] = h[n].a_dst
// ---------------------------------------------------------------------------
__global__ void attn_dot(const float* __restrict__ Hm,
                         const float* __restrict__ avs,
                         const float* __restrict__ avd,
                         float* __restrict__ vs, float* __restrict__ vd, int F) {
  int node = blockIdx.x * (blockDim.x >> 5) + (threadIdx.x >> 5);
  if (node >= N_NODES) return;
  int lane = threadIdx.x & 31;
  const float* hrow = Hm + (size_t)node * F;
  float s = 0.f, d = 0.f;
  for (int f = lane; f < F; f += 32) {
    float h = hrow[f];
    s += h * avs[f];
    d += h * avd[f];
  }
#pragma unroll
  for (int off = 16; off > 0; off >>= 1) {
    s += __shfl_xor(s, off, 32);
    d += __shfl_xor(d, off, 32);
  }
  if (lane == 0) { vs[node] = s; vd[node] = d; }
}

__global__ void fill_u32(unsigned* __restrict__ p, unsigned v, int n) {
  int i = blockIdx.x * blockDim.x + threadIdx.x;
  if (i < n) p[i] = v;
}
__global__ void fill_f32(float* __restrict__ p, float v, int n) {
  int i = blockIdx.x * blockDim.x + threadIdx.x;
  if (i < n) p[i] = v;
}

// per-edge logit + segment max (atomic max on order-preserving uint encoding)
__global__ void edge_logit(const int* __restrict__ ei,
                           const float* __restrict__ vs,
                           const float* __restrict__ vd,
                           float* __restrict__ e, unsigned* __restrict__ menc) {
  int i = blockIdx.x * blockDim.x + threadIdx.x;
  if (i >= ET) return;
  int s, d;
  if (i < N_EDGES) { s = ei[i]; d = ei[N_EDGES + i]; }
  else             { s = d = i - N_EDGES; }
  float x = vs[s] + vd[d];
  x = (x > 0.f) ? x : NEG_SLOPE * x;          // leaky relu
  e[i] = x;
  atomicMax(&menc[d], fenc(x));
}

// exp(e - max) + segment sum
__global__ void edge_exp(const int* __restrict__ ei, float* __restrict__ e,
                         const unsigned* __restrict__ menc,
                         float* __restrict__ denom) {
  int i = blockIdx.x * blockDim.x + threadIdx.x;
  if (i >= ET) return;
  int d = (i < N_EDGES) ? ei[N_EDGES + i] : i - N_EDGES;
  float ex = __expf(e[i] - fdec(menc[d]));
  e[i] = ex;
  atomicAdd(&denom[d], ex);
}

// weighted scatter-aggregate: out[dst] += alpha * h[src]; wave per edge.
// Lane-contiguous float4 gathers (one b128 per lane; F=128 -> single pass),
// atomics resolve in L2 (h is 25.6MB << 192MB L2).
__global__ void edge_aggregate(const int* __restrict__ ei,
                               const float* __restrict__ e,
                               const float* __restrict__ denom,
                               const float* __restrict__ Hm,
                               float* __restrict__ out, int F) {
  int edge = blockIdx.x * (blockDim.x >> 5) + (threadIdx.x >> 5);
  if (edge >= ET) return;
  int lane = threadIdx.x & 31;
  int s, d;
  if (edge < N_EDGES) { s = ei[edge]; d = ei[N_EDGES + edge]; }
  else                { s = d = edge - N_EDGES; }
  float alpha = e[edge] / (denom[d] + 1e-16f);
  const float* hrow = Hm + (size_t)s * F;
  float* orow = out + (size_t)d * F;
  for (int f0 = lane * 4; f0 < F; f0 += 128) {
    float4 hv = *(const float4*)(hrow + f0);   // F is a multiple of 4 (128 or 40)
    atomicAdd(&orow[f0 + 0], alpha * hv.x);
    atomicAdd(&orow[f0 + 1], alpha * hv.y);
    atomicAdd(&orow[f0 + 2], alpha * hv.z);
    atomicAdd(&orow[f0 + 3], alpha * hv.w);
  }
}

// BatchNorm (eval) + ReLU, in place, F == H_DIM (power of two)
__global__ void bn_relu(float* __restrict__ h,
                        const float* __restrict__ g, const float* __restrict__ b,
                        const float* __restrict__ m, const float* __restrict__ v) {
  int i = blockIdx.x * blockDim.x + threadIdx.x;
  if (i >= N_NODES * H_DIM) return;
  int f = i & (H_DIM - 1);
  float inv = rsqrtf(v[f] + BN_EPS);
  float val = (h[i] - m[f]) * g[f] * inv + b[f];
  h[i] = val > 0.f ? val : 0.f;
}

extern "C" void kernel_launch(void* const* d_in, const int* in_sizes, int n_in,
                              void* d_out, int out_size, void* d_ws, size_t ws_size,
                              hipStream_t stream) {
  (void)in_sizes; (void)n_in; (void)out_size; (void)ws_size;
  const float* x  = (const float*)d_in[0];
  const int*   ei = (const int*)d_in[1];   // [2, E] row-major: src then dst
  const float* W[4]  = {(const float*)d_in[2], (const float*)d_in[3],
                        (const float*)d_in[4], (const float*)d_in[5]};
  const float* AS[4] = {(const float*)d_in[6], (const float*)d_in[7],
                        (const float*)d_in[8], (const float*)d_in[9]};
  const float* AD[4] = {(const float*)d_in[10], (const float*)d_in[11],
                        (const float*)d_in[12], (const float*)d_in[13]};
  const float* bn_g = (const float*)d_in[14];
  const float* bn_b = (const float*)d_in[15];
  const float* bn_m = (const float*)d_in[16];
  const float* bn_v = (const float*)d_in[17];

  // workspace carve-up (floats)
  float* p = (float*)d_ws;
  float* h1 = p;            p += (size_t)N_NODES * H_DIM;   // gemm result (hidden)
  float* h2 = p;            p += (size_t)N_NODES * H_DIM;   // aggregate result (hidden)
  float* hg = p;            p += (size_t)N_NODES * C_DIM;   // gemm result (final)
  float* vs = p;            p += N_NODES;
  float* vd = p;            p += N_NODES;
  unsigned* menc = (unsigned*)p; p += N_NODES;
  float* denom = p;         p += N_NODES;
  float* ebuf = p;          p += ET;
  // packed bf16 weight fragments
  __bf16* wp = (__bf16*)p;
  __bf16* Wp[4];
  const int Ks[4]  = {F_INPUT, H_DIM, H_DIM, H_DIM};
  const int Ms[4]  = {H_DIM,   H_DIM, H_DIM, C_DIM};
  for (int i = 0; i < 4; ++i) {
    Wp[i] = wp;
    wp += (size_t)(Ks[i] >> 5) * ((Ms[i] + 15) / 16) * 512;
  }

  // one-time weight packing (cheap: <= 32K elements per layer)
  for (int i = 0; i < 4; ++i) {
    int colTiles = (Ms[i] + 15) / 16;
    int total = (Ks[i] >> 5) * colTiles * 512;
    pack_w_bf16<<<(total + 255) / 256, 256, 0, stream>>>(W[i], Wp[i], Ks[i], Ms[i], colTiles);
  }

  const float* cur = x;
  int Fin = F_INPUT;

  for (int layer = 0; layer < 4; ++layer) {
    const int Fout = (layer == 3) ? C_DIM : H_DIM;
    float* gem = (layer == 3) ? hg : h1;
    float* out = (layer == 3) ? (float*)d_out : h2;

    // dense projection via WMMA: 4 waves/block, each owns a 16-row strip
    if (layer == 3)
      gemm_wmma_strip<3><<<(ROW_TILES + 3) / 4, 128, 0, stream>>>(cur, Wp[layer], gem, Fin, Fout);
    else
      gemm_wmma_strip<8><<<(ROW_TILES + 3) / 4, 128, 0, stream>>>(cur, Wp[layer], gem, Fin, Fout);

    // attention projections
    attn_dot<<<(N_NODES + 7) / 8, 256, 0, stream>>>(gem, AS[layer], AD[layer], vs, vd, Fout);
    // init segment buffers + output accumulator
    fill_u32<<<(N_NODES + 255) / 256, 256, 0, stream>>>(menc, 0x007FFFFFu /* fenc(-inf) */, N_NODES);
    fill_f32<<<(N_NODES + 255) / 256, 256, 0, stream>>>(denom, 0.f, N_NODES);
    fill_f32<<<(N_NODES * Fout + 255) / 256, 256, 0, stream>>>(out, 0.f, N_NODES * Fout);
    // segment softmax + aggregate
    edge_logit<<<(ET + 255) / 256, 256, 0, stream>>>(ei, vs, vd, ebuf, menc);
    edge_exp<<<(ET + 255) / 256, 256, 0, stream>>>(ei, ebuf, menc, denom);
    edge_aggregate<<<(ET + 7) / 8, 256, 0, stream>>>(ei, ebuf, denom, gem, out, Fout);

    if (layer < 3) {
      bn_relu<<<(N_NODES * H_DIM + 255) / 256, 256, 0, stream>>>(
          out, bn_g + layer * H_DIM, bn_b + layer * H_DIM,
          bn_m + layer * H_DIM, bn_v + layer * H_DIM);
      cur = out;        // h2 feeds the next layer; h1/hg are free to overwrite
      Fin = H_DIM;
    }
  }
}
